// SFT_16844861735534
// MI455X (gfx1250) — matrix operand order repaired and checked by reading, not compiled, not run on hardware
//
#include <hip/hip_runtime.h>

// Fused flash-attention implementation of:
//   emb_n = l2norm_rows(emb);  W = emb_n emb_n^T / 0.1;  T = softmax(W); out = T @ emb
// Q=K = sqrt(10)*emb_n (bf16), V = emb (bf16, stored transposed), f32 accumulation.
// K tiles are staged into LDS by the Tensor Data Mover (double-buffered, async),
// with TDM row padding (512B rows + 16B pad) for conflict-free ds_load_b128.

typedef __attribute__((ext_vector_type(16))) __bf16 v16bf;
typedef __attribute__((ext_vector_type(8)))  __bf16 v8bf;
typedef __attribute__((ext_vector_type(8)))  float  v8f;
typedef __attribute__((ext_vector_type(4)))  float  v4f;
typedef __attribute__((ext_vector_type(4)))  unsigned int v4ui;
typedef __attribute__((ext_vector_type(8)))  int v8si;
typedef __attribute__((ext_vector_type(4)))  int v4si;

#define NROWS 8192
#define DIM   256
#define KT    32           // keys per tile
#define WAVES 2            // waves per block, 16 query rows per wave
#define KROW  264          // padded K-tile LDS row stride in bf16 elems (528 B = 132 dw)

#if defined(__has_builtin)
#if __has_builtin(__builtin_amdgcn_tensor_load_to_lds)
#define HAVE_TDM 1
#endif
#endif

union U16bf { v16bf v; v8bf h2[2]; };

// ---------------- Phase 0: normalize + cast + transpose ----------------
__global__ __launch_bounds__(256)
void sft_norm_kernel(const float* __restrict__ emb,
                     __bf16* __restrict__ qbf,      // [NROWS][DIM]  sqrt(10)*emb_n
                     __bf16* __restrict__ vt) {     // [DIM][NROWS]  emb^T
  const int wave = threadIdx.x >> 5;
  const int lane = threadIdx.x & 31;
  const int row  = blockIdx.x * 8 + wave;
  const float* src = emb + row * DIM + lane * 8;
  v4f a = ((const v4f*)src)[0];
  v4f b = ((const v4f*)src)[1];
  float ss = a.x*a.x + a.y*a.y + a.z*a.z + a.w*a.w
           + b.x*b.x + b.y*b.y + b.z*b.z + b.w*b.w;
  #pragma unroll
  for (int m = 16; m >= 1; m >>= 1) ss += __shfl_xor(ss, m, 32);
  float nrm = fmaxf(sqrtf(ss), 1e-12f);
  const float s = 3.16227766016838f / nrm;   // sqrt(1/sigma) folded symmetrically
  float vv[8] = {a.x, a.y, a.z, a.w, b.x, b.y, b.z, b.w};
  #pragma unroll
  for (int i = 0; i < 8; ++i) {
    int d = lane * 8 + i;
    qbf[row * DIM + d]  = (__bf16)(vv[i] * s);
    vt[d * NROWS + row] = (__bf16)vv[i];
  }
}

#ifdef HAVE_TDM
// 2D TDM load: tile_d1 rows of tile_d0 elements (2-byte), row stride d0_stride
// elements in memory, optional LDS padding (pad fields per D# encoding).
__device__ __attribute__((always_inline))
void tdm_load_2d(unsigned lds_off, const void* gaddr,
                 unsigned tensor_d0, unsigned tensor_d1,
                 unsigned tile_d0, unsigned tile_d1,
                 unsigned long long d0_stride,
                 unsigned pad_en, unsigned pad_ivl, unsigned pad_amt) {
  unsigned long long ga = (unsigned long long)gaddr;
  v4ui g0;
  g0[0] = 1u;                                      // count=1, no gather
  g0[1] = lds_off;                                 // lds_addr (bytes)
  g0[2] = (unsigned)ga;                            // global_addr[31:0]
  g0[3] = (unsigned)(ga >> 32) | (2u << 30);       // global_addr[56:32] | type=2
  v8si g1;
  g1[0] = (int)((1u << 16) | (pad_en << 20) | (pad_ivl << 22) | (pad_amt << 25));
  g1[1] = (int)((tensor_d0 & 0xFFFFu) << 16);                       // tdim0[15:0]
  g1[2] = (int)((tensor_d0 >> 16) | ((tensor_d1 & 0xFFFFu) << 16)); // tdim0 hi | tdim1 lo
  g1[3] = (int)((tensor_d1 >> 16) | (tile_d0 << 16));               // tdim1 hi | tile_d0
  g1[4] = (int)(tile_d1 & 0xFFFFu);                                 // tile_d1, tile_d2=0
  g1[5] = (int)(unsigned)d0_stride;                                 // dim0_stride[31:0]
  g1[6] = (int)(unsigned)(d0_stride >> 32);                         // dim0_stride[47:32]
  g1[7] = 0;
  v4si gz4 = {};
  v8si gz8 = {};
  // clang-23 / therock-10.0 6-arg form: (g0, g1, g2, g3, g_extra, cpol)
  __builtin_amdgcn_tensor_load_to_lds(g0, g1, gz4, gz4, gz8, 0);
}
#endif

// ---------------- Phase 1: fused attention ----------------
__global__ __launch_bounds__(WAVES * 32)
void sft_attn_kernel(const __bf16* __restrict__ qbf,
                     const __bf16* __restrict__ vt,
                     float* __restrict__ out) {
  __shared__ __bf16 KL[2][KT * KROW];          // double-buffered K tile (2x16.9 KB)
  __shared__ __bf16 plds[WAVES][16 * 32];      // per-wave P transpose staging
  const int wave = threadIdx.x >> 5;
  const int lane = threadIdx.x & 31;
  const int h    = lane >> 4;                  // lane half (K-offset select)
  const int nl   = lane & 15;                  // position within half
  const int row_base = blockIdx.x * (WAVES * 16) + wave * 16;
  const int NT = NROWS / KT;

  // Q fragments: A-layout 16x32 bf16, 8 k-chunks. Lane(nl,h) holds row M=nl;
  // v0..7 = K[16h..16h+15] (contiguous).
  v16bf q[8];
  {
    const __bf16* qrow = qbf + (row_base + nl) * DIM;
    #pragma unroll
    for (int kc = 0; kc < 8; ++kc) {
      U16bf u;
      u.h2[0] = *(const v8bf*)(qrow + kc * 32 + 16 * h);
      u.h2[1] = *(const v8bf*)(qrow + kc * 32 + 16 * h + 8);
      q[kc] = u.v;
    }
  }

  v8f zero = {};
  v8f o[16];
  #pragma unroll
  for (int i = 0; i < 16; ++i) o[i] = zero;
  float mrow[8], lrow[8];
  #pragma unroll
  for (int r = 0; r < 8; ++r) { mrow[r] = -1e30f; lrow[r] = 0.f; }

  __bf16* myp = &plds[wave][0];

#ifdef HAVE_TDM
  // Preload tile 0 via Tensor Data Mover (wave 0 issues; TENSORcnt is per-wave).
  if (wave == 0) {
    tdm_load_2d((unsigned)(size_t)&KL[0][0], qbf,
                DIM, NROWS, DIM, KT, DIM,
                1u, 6u, 3u);                  // pad 4 dwords per 128 dwords -> 528B rows
    __builtin_amdgcn_s_wait_tensorcnt(0);
  }
  __syncthreads();
#endif

  for (int kt = 0; kt < NT; ++kt) {
    const int kb  = kt * KT;
    const int cur = kt & 1;

#ifdef HAVE_TDM
    if (wave == 0 && kt + 1 < NT) {           // async-prefetch next K tile
      tdm_load_2d((unsigned)(size_t)&KL[cur ^ 1][0],
                  qbf + (size_t)(kb + KT) * DIM,
                  DIM, NROWS, DIM, KT, DIM, 1u, 6u, 3u);
    }
    const __bf16* kl = &KL[cur][0];
#endif

    // ---- S = Q K^T (two 16-key n-subtiles); B frags from LDS-staged K tile ----
    v8f s0 = zero, s1 = zero;
    #pragma unroll
    for (int kc = 0; kc < 8; ++kc) {
      U16bf b0, b1;
#ifdef HAVE_TDM
      b0.h2[0] = *(const v8bf*)(kl + nl * KROW + kc * 32 + 16 * h);
      b0.h2[1] = *(const v8bf*)(kl + nl * KROW + kc * 32 + 16 * h + 8);
      b1.h2[0] = *(const v8bf*)(kl + (nl + 16) * KROW + kc * 32 + 16 * h);
      b1.h2[1] = *(const v8bf*)(kl + (nl + 16) * KROW + kc * 32 + 16 * h + 8);
#else
      const __bf16* krow0 = qbf + (kb + nl) * DIM + 16 * h;
      b0.h2[0] = *(const v8bf*)(krow0 + kc * 32);
      b0.h2[1] = *(const v8bf*)(krow0 + kc * 32 + 8);
      b1.h2[0] = *(const v8bf*)(krow0 + 16 * DIM + kc * 32);
      b1.h2[1] = *(const v8bf*)(krow0 + 16 * DIM + kc * 32 + 8);
#endif
      s0 = __builtin_amdgcn_wmma_f32_16x16x32_bf16(false, q[kc], false, b0.v,
                                                   (short)0, s0, false, false);
      s1 = __builtin_amdgcn_wmma_f32_16x16x32_bf16(false, q[kc], false, b1.v,
                                                   (short)0, s1, false, false);
    }

    // ---- online softmax (row r+8h lives in VGPR r across 16 lanes of half h) ----
    float p0[8], p1[8];
    #pragma unroll
    for (int r = 0; r < 8; ++r) {
      float t = fmaxf(s0[r], s1[r]);
      #pragma unroll
      for (int m = 1; m < 16; m <<= 1) t = fmaxf(t, __shfl_xor(t, m, 32));
      float mn = fmaxf(mrow[r], t);
      float rs = __expf(mrow[r] - mn);
      mrow[r] = mn;
      float e0 = __expf(s0[r] - mn);
      float e1 = __expf(s1[r] - mn);
      float sum = e0 + e1;
      #pragma unroll
      for (int m = 1; m < 16; m <<= 1) sum += __shfl_xor(sum, m, 32);
      lrow[r] = lrow[r] * rs + sum;
      p0[r] = e0; p1[r] = e1;
      #pragma unroll
      for (int ds = 0; ds < 16; ++ds) o[ds][r] *= rs;
    }

    // ---- P: C-layout -> A-layout via per-wave LDS tile ----
    #pragma unroll
    for (int r = 0; r < 8; ++r) {
      int pm = r + 8 * h;
      myp[pm * 32 + nl]      = (__bf16)p0[r];
      myp[pm * 32 + nl + 16] = (__bf16)p1[r];
    }
    __builtin_amdgcn_fence(__ATOMIC_ACQ_REL, "wavefront");
    U16bf pa;
    pa.h2[0] = *(const v8bf*)(myp + nl * 32 + 16 * h);
    pa.h2[1] = *(const v8bf*)(myp + nl * 32 + 16 * h + 8);

    // ---- O += P V : B frag = 16 contiguous bf16 from transposed V (global) ----
    const __bf16* vbase = vt + kb + 16 * h;
    #pragma unroll
    for (int ds = 0; ds < 16; ++ds) {
      U16bf bv;
      bv.h2[0] = *(const v8bf*)(vbase + (ds * 16 + nl) * NROWS);
      bv.h2[1] = *(const v8bf*)(vbase + (ds * 16 + nl) * NROWS + 8);
      o[ds] = __builtin_amdgcn_wmma_f32_16x16x32_bf16(false, pa.v, false, bv.v,
                                                      (short)0, o[ds], false, false);
    }

#ifdef HAVE_TDM
    if (wave == 0) __builtin_amdgcn_s_wait_tensorcnt(0);
    __syncthreads();                          // publish next K tile, retire current
#endif
  }

  // ---- epilogue: divide by row sums, store f32 ----
  float inv[8];
  #pragma unroll
  for (int r = 0; r < 8; ++r) inv[r] = 1.0f / lrow[r];
  #pragma unroll
  for (int ds = 0; ds < 16; ++ds) {
    #pragma unroll
    for (int r = 0; r < 8; ++r) {
      out[(row_base + r + 8 * h) * DIM + ds * 16 + nl] = o[ds][r] * inv[r];
    }
  }
}

extern "C" void kernel_launch(void* const* d_in, const int* in_sizes, int n_in,
                              void* d_out, int out_size, void* d_ws, size_t ws_size,
                              hipStream_t stream) {
  const float* emb = (const float*)d_in[0];
  float* out = (float*)d_out;
  __bf16* qbf = (__bf16*)d_ws;                     // 4 MB
  __bf16* vt  = qbf + (size_t)NROWS * DIM;         // 4 MB
  sft_norm_kernel<<<NROWS / 8, 256, 0, stream>>>(emb, qbf, vt);
  sft_attn_kernel<<<NROWS / (WAVES * 16), WAVES * 32, 0, stream>>>(qbf, vt, out);
}
